// DecoderLayer_33277406609814
// MI455X (gfx1250) — matrix operand order repaired and checked by reading, not compiled, not run on hardware
//
#include <hip/hip_runtime.h>
#include <hip/hip_bf16.h>

#define TOK 4096   // B*S tokens
#define DM  1024
#define DF  4096
#define SEQ 2048
#define NB  2
#define NH  16

typedef __attribute__((ext_vector_type(16))) _Float16 v16h;
typedef __attribute__((ext_vector_type(8)))  float    v8f;
typedef unsigned int u32;

union U32H2 { u32 u; _Float16 h[2]; };

__device__ __forceinline__ v8f wmma16(v16h a, v16h b, v8f c) {
  // D = A(16x32 f16) * B(32x16 f16) + C(16x16 f32)
  return __builtin_amdgcn_wmma_f32_16x16x32_f16(false, a, false, b, (short)0, c, false, false);
}

// A fragment (16x32 f16, M x K): rp = &src[(row0 + (lane&15))*ld + k0]
// lane group g: elements 0..7 -> k = g*8 + i ; elements 8..15 -> k = 16 + g*8 + (i-8)
__device__ __forceinline__ v16h ld_frag_a(const _Float16* rp, int g) {
  v16h r;
  ((uint4*)&r)[0] = *(const uint4*)(rp + g * 8);
  ((uint4*)&r)[1] = *(const uint4*)(rp + 16 + g * 8);
  return r;
}
// B fragment (32x16 f16, K x N) from K-major storage Bt[n][k]:
// rp = &Bt[(n0 + (lane&15))*ldk + k0]; lane group g: elements 0..7 -> k = g*8 + i ; elements 8..15 -> k = 16 + g*8 + (i-8)
__device__ __forceinline__ v16h ld_frag_b(const _Float16* rp, int g) {
  v16h r;
  ((uint4*)&r)[0] = *(const uint4*)(rp + g * 8);
  ((uint4*)&r)[1] = *(const uint4*)(rp + 16 + g * 8);
  return r;
}

__global__ void f32_to_f16_kernel(const float* __restrict__ s, _Float16* __restrict__ d, int n) {
  int i = blockIdx.x * blockDim.x + threadIdx.x;
  if (i < n) d[i] = (_Float16)s[i];
}

// C[M,N] = A[M,K](f16,row-major) @ Bw[K,N](f16,row-major) + bias (+ReLU).
// 256 threads = 8 waves in a 4(M) x 2(N) grid; 128x128 tile; K-step 32.
// Each wave: 32x64 output = 2x4 accumulators, 8 WMMAs per K-step.
template <bool RELU, bool W32, bool W16>
__global__ __launch_bounds__(256) void gemm_wmma_kernel(
    const _Float16* __restrict__ A, const _Float16* __restrict__ Bw,
    const float* __restrict__ bias, float* __restrict__ C32,
    _Float16* __restrict__ C16, int M, int N, int K)
{
  __shared__ _Float16 As[128][40];   // [m][k], row stride 80B (16B aligned)
  __shared__ _Float16 Bs[128][40];   // [n][k] (transposed tile)
  const int tid = threadIdx.x;
  const int wv = tid >> 5, lane = tid & 31;
  const int g = lane >> 4, ln = lane & 15;
  const int wm = wv >> 1, wn = wv & 1;
  const int m0 = blockIdx.y * 128, n0 = blockIdx.x * 128;
  v8f acc[2][4] = {};
  for (int kk = 0; kk < K; kk += 32) {
    // A tile: 128x32 halves = 512 x b128, coalesced
#pragma unroll
    for (int i = 0; i < 2; ++i) {
      int e = tid + i * 256;
      int r = e >> 2, q = e & 3;      // 4 x uint4 per 32-half row
      ((uint4*)&As[r][0])[q] = ((const uint4*)(A + (size_t)(m0 + r) * K + kk))[q];
    }
    // B tile: read row-major [k][n] (32x128) as b128, store transposed [n][k]
#pragma unroll
    for (int i = 0; i < 2; ++i) {
      int e = tid + i * 256;
      int kr = e >> 4, q = e & 15;    // 16 x uint4 per 128-half row
      uint4 val = ((const uint4*)(Bw + (size_t)(kk + kr) * N + n0))[q];
      U32H2 c0, c1, c2, c3;
      c0.u = val.x; c1.u = val.y; c2.u = val.z; c3.u = val.w;
      int nb = 8 * q;
      Bs[nb + 0][kr] = c0.h[0]; Bs[nb + 1][kr] = c0.h[1];
      Bs[nb + 2][kr] = c1.h[0]; Bs[nb + 3][kr] = c1.h[1];
      Bs[nb + 4][kr] = c2.h[0]; Bs[nb + 5][kr] = c2.h[1];
      Bs[nb + 6][kr] = c3.h[0]; Bs[nb + 7][kr] = c3.h[1];
    }
    __syncthreads();
    v16h a0 = ld_frag_a(&As[wm * 32 + ln][0], g);
    v16h a1 = ld_frag_a(&As[wm * 32 + 16 + ln][0], g);
#pragma unroll
    for (int nt = 0; nt < 4; ++nt) {
      v16h b = ld_frag_b(&Bs[wn * 64 + nt * 16 + ln][0], g);
      acc[0][nt] = wmma16(a0, b, acc[0][nt]);
      acc[1][nt] = wmma16(a1, b, acc[1][nt]);
    }
    __syncthreads();
  }
#pragma unroll
  for (int nt = 0; nt < 4; ++nt) {
    int n = n0 + wn * 64 + nt * 16 + ln;
    float bv = bias[n];
#pragma unroll
    for (int mt = 0; mt < 2; ++mt) {
#pragma unroll
      for (int v = 0; v < 8; ++v) {
        float val = acc[mt][nt][v] + bv;
        if (RELU) val = fmaxf(val, 0.0f);
        size_t idx = (size_t)(m0 + wm * 32 + mt * 16 + v + 8 * g) * N + n;
        if (W32) C32[idx] = val;
        if (W16) C16[idx] = (_Float16)val;
      }
    }
  }
}

// Flash attention, one block = (b, h, 64 query rows), 4 waves x 16 rows.
// Q/K/V: f16 [NB*SEQ, DM], head h at column h*64, depth 64. O16: f16 context.
__global__ __launch_bounds__(128) void flash_attn_kernel(
    const _Float16* __restrict__ Q, const _Float16* __restrict__ Kh,
    const _Float16* __restrict__ Vh, const float* __restrict__ pad,
    _Float16* __restrict__ O16, int causal)
{
  __shared__ _Float16 Kt[32][72];      // [key][depth 0..63], stride 144B (16B aligned)
  __shared__ _Float16 Vt[64][40];      // [depth][key 0..31]  (transposed)
  __shared__ _Float16 Ps[4][16][40];   // per-wave P staging [m][key]
  const int tid = threadIdx.x;
  const int wv = tid >> 5, lane = tid & 31;
  const int g = lane >> 4, ln = lane & 15;
  const int b = blockIdx.z, h = blockIdx.y, qb = blockIdx.x;
  const int q0 = qb * 64 + wv * 16;
  const size_t rowQ = (size_t)(b * SEQ + q0 + ln) * DM + h * 64;
  v16h aq0, aq1;                       // Q A-frags, depth 0..31 and 32..63
  ((uint4*)&aq0)[0] = *(const uint4*)(Q + rowQ + g * 8);
  ((uint4*)&aq0)[1] = *(const uint4*)(Q + rowQ + 16 + g * 8);
  ((uint4*)&aq1)[0] = *(const uint4*)(Q + rowQ + 32 + g * 8);
  ((uint4*)&aq1)[1] = *(const uint4*)(Q + rowQ + 48 + g * 8);
  v8f o0 = {}, o1 = {}, o2 = {}, o3 = {};
  float mrow[8], lrow[8];
#pragma unroll
  for (int v = 0; v < 8; ++v) { mrow[v] = -1e30f; lrow[v] = 0.0f; }
  const int nkb = causal ? (qb * 64 + 64) / 32 : SEQ / 32;   // block-uniform
  for (int j = 0; j < nkb; ++j) {
    const int kb = j * 32;
    // cooperative K/V tile load (32 keys x 64 depth each), b128 global loads
#pragma unroll
    for (int i = 0; i < 2; ++i) {
      int e = tid + i * 128;
      int kr = e >> 3, q = e & 7;     // 8 x uint4 per 64-half row
      size_t src = (size_t)(b * SEQ + kb + kr) * DM + h * 64;
      ((uint4*)&Kt[kr][0])[q] = ((const uint4*)(Kh + src))[q];
      uint4 vv = ((const uint4*)(Vh + src))[q];
      U32H2 c0, c1, c2, c3;
      c0.u = vv.x; c1.u = vv.y; c2.u = vv.z; c3.u = vv.w;
      int d0 = 8 * q;
      Vt[d0 + 0][kr] = c0.h[0]; Vt[d0 + 1][kr] = c0.h[1];
      Vt[d0 + 2][kr] = c1.h[0]; Vt[d0 + 3][kr] = c1.h[1];
      Vt[d0 + 4][kr] = c2.h[0]; Vt[d0 + 5][kr] = c2.h[1];
      Vt[d0 + 6][kr] = c3.h[0]; Vt[d0 + 7][kr] = c3.h[1];
    }
    __syncthreads();
    // logits S = Q Kh^T : two 16x16 frags (keys 0..15, 16..31)
    v8f s0 = {}, s1 = {};
    s0 = wmma16(aq0, ld_frag_b(&Kt[ln][0], g), s0);
    s0 = wmma16(aq1, ld_frag_b(&Kt[ln][32], g), s0);
    s1 = wmma16(aq0, ld_frag_b(&Kt[16 + ln][0], g), s1);
    s1 = wmma16(aq1, ld_frag_b(&Kt[16 + ln][32], g), s1);
    const int key0 = kb + ln, key1 = kb + 16 + ln;
    float pm0 = 0.0f, pm1 = 0.0f;
    if (pad) { pm0 = pad[b * SEQ + key0]; pm1 = pad[b * SEQ + key1]; }
#pragma unroll
    for (int v = 0; v < 8; ++v) {
      int qr = q0 + v + 8 * g;
      float x0 = s0[v] * 0.125f - 1e9f * pm0;
      float x1 = s1[v] * 0.125f - 1e9f * pm1;
      if (causal) {
        x0 += (key0 > qr) ? -1e9f : 0.0f;
        x1 += (key1 > qr) ? -1e9f : 0.0f;
      }
      // row reduction across the 16 lanes holding this row
      float rm = fmaxf(x0, x1);
      rm = fmaxf(rm, __shfl_xor(rm, 1, 32));
      rm = fmaxf(rm, __shfl_xor(rm, 2, 32));
      rm = fmaxf(rm, __shfl_xor(rm, 4, 32));
      rm = fmaxf(rm, __shfl_xor(rm, 8, 32));
      float mn = fmaxf(mrow[v], rm);
      float c  = __expf(mrow[v] - mn);
      mrow[v] = mn;
      float p0 = __expf(x0 - mn);
      float p1 = __expf(x1 - mn);
      float rs = p0 + p1;
      rs += __shfl_xor(rs, 1, 32);
      rs += __shfl_xor(rs, 2, 32);
      rs += __shfl_xor(rs, 4, 32);
      rs += __shfl_xor(rs, 8, 32);
      lrow[v] = lrow[v] * c + rs;
      o0[v] *= c; o1[v] *= c; o2[v] *= c; o3[v] *= c;
      Ps[wv][v + 8 * g][ln]      = (_Float16)p0;   // C-layout -> [m][key]
      Ps[wv][v + 8 * g][16 + ln] = (_Float16)p1;
    }
    asm volatile("s_wait_dscnt 0" ::: "memory");   // wave-local LDS RAW
    v16h ap = ld_frag_a(&Ps[wv][ln][0], g);        // P as A-frag (16x32)
    o0 = wmma16(ap, ld_frag_b(&Vt[ln][0], g), o0);
    o1 = wmma16(ap, ld_frag_b(&Vt[16 + ln][0], g), o1);
    o2 = wmma16(ap, ld_frag_b(&Vt[32 + ln][0], g), o2);
    o3 = wmma16(ap, ld_frag_b(&Vt[48 + ln][0], g), o3);
    __syncthreads();
  }
#pragma unroll
  for (int v = 0; v < 8; ++v) {
    float inv = 1.0f / lrow[v];
    size_t row = (size_t)(b * SEQ + q0 + v + 8 * g) * DM + h * 64 + ln;
    O16[row]      = (_Float16)(o0[v] * inv);
    O16[row + 16] = (_Float16)(o1[v] * inv);
    O16[row + 32] = (_Float16)(o2[v] * inv);
    O16[row + 48] = (_Float16)(o3[v] * inv);
  }
}

// Y = LayerNorm(X + R) * G + Bb ; one row of 1024 per 256-thread block.
template <bool W16>
__global__ __launch_bounds__(256) void layernorm_kernel(
    const float* __restrict__ X, const float* __restrict__ R,
    const float* __restrict__ G, const float* __restrict__ Bb,
    float* __restrict__ Y32, _Float16* __restrict__ Y16)
{
  const int row = blockIdx.x, tid = threadIdx.x;
  __shared__ float red[8];
  const size_t base = (size_t)row * DM;
  float xv[4];
  float s = 0.0f;
#pragma unroll
  for (int i = 0; i < 4; ++i) {
    int c = tid + i * 256;
    xv[i] = X[base + c] + R[base + c];
    s += xv[i];
  }
#pragma unroll
  for (int m = 16; m >= 1; m >>= 1) s += __shfl_xor(s, m, 32);
  if ((tid & 31) == 0) red[tid >> 5] = s;
  __syncthreads();
  float tot = 0.0f;
#pragma unroll
  for (int w = 0; w < 8; ++w) tot += red[w];
  const float mu = tot * (1.0f / DM);
  __syncthreads();
  float vs = 0.0f;
#pragma unroll
  for (int i = 0; i < 4; ++i) { float d = xv[i] - mu; vs += d * d; }
#pragma unroll
  for (int m = 16; m >= 1; m >>= 1) vs += __shfl_xor(vs, m, 32);
  if ((tid & 31) == 0) red[tid >> 5] = vs;
  __syncthreads();
  float tv = 0.0f;
#pragma unroll
  for (int w = 0; w < 8; ++w) tv += red[w];
  const float rstd = rsqrtf(tv * (1.0f / DM) + 1e-6f);
#pragma unroll
  for (int i = 0; i < 4; ++i) {
    int c = tid + i * 256;
    float y = (xv[i] - mu) * rstd * G[c] + Bb[c];
    Y32[base + c] = y;
    if (W16) Y16[base + c] = (_Float16)y;
  }
}

extern "C" void kernel_launch(void* const* d_in, const int* in_sizes, int n_in,
                              void* d_out, int out_size, void* d_ws, size_t ws_size,
                              hipStream_t stream) {
  (void)in_sizes; (void)n_in; (void)out_size; (void)ws_size;
  const float* x    = (const float*)d_in[0];
  const float* enc  = (const float*)d_in[1];
  const float* padm = (const float*)d_in[3];        // [B,1,1,S] -> [B*S]
  const float* wq1 = (const float*)d_in[4];  const float* bq1 = (const float*)d_in[5];
  const float* wk1 = (const float*)d_in[6];  const float* bk1 = (const float*)d_in[7];
  const float* wv1 = (const float*)d_in[8];  const float* bv1 = (const float*)d_in[9];
  const float* wo1 = (const float*)d_in[10]; const float* bo1 = (const float*)d_in[11];
  const float* wq2 = (const float*)d_in[12]; const float* bq2 = (const float*)d_in[13];
  const float* wk2 = (const float*)d_in[14]; const float* bk2 = (const float*)d_in[15];
  const float* wv2 = (const float*)d_in[16]; const float* bv2 = (const float*)d_in[17];
  const float* wo2 = (const float*)d_in[18]; const float* bo2 = (const float*)d_in[19];
  const float* fw1 = (const float*)d_in[20]; const float* fb1 = (const float*)d_in[21];
  const float* fw2 = (const float*)d_in[22]; const float* fb2 = (const float*)d_in[23];
  const float* g1  = (const float*)d_in[24]; const float* be1 = (const float*)d_in[25];
  const float* g2  = (const float*)d_in[26]; const float* be2 = (const float*)d_in[27];
  const float* g3  = (const float*)d_in[28]; const float* be3 = (const float*)d_in[29];

  char* base = (char*)d_ws; size_t off = 0;
  auto alloc16 = [&](size_t elems) -> _Float16* {
    _Float16* p = (_Float16*)(base + off);
    off = (off + elems * sizeof(_Float16) + 255) & ~(size_t)255;
    return p;
  };
  auto alloc32 = [&](size_t elems) -> float* {
    float* p = (float*)(base + off);
    off = (off + elems * sizeof(float) + 255) & ~(size_t)255;
    return p;
  };

  _Float16 *x16 = alloc16((size_t)TOK * DM), *e16 = alloc16((size_t)TOK * DM);
  _Float16 *hq1 = alloc16((size_t)DM * DM), *hk1 = alloc16((size_t)DM * DM);
  _Float16 *hv1 = alloc16((size_t)DM * DM), *ho1 = alloc16((size_t)DM * DM);
  _Float16 *hq2 = alloc16((size_t)DM * DM), *hk2 = alloc16((size_t)DM * DM);
  _Float16 *hv2 = alloc16((size_t)DM * DM), *ho2 = alloc16((size_t)DM * DM);
  _Float16 *hw1 = alloc16((size_t)DM * DF), *hw2 = alloc16((size_t)DF * DM);
  _Float16 *q16 = alloc16((size_t)TOK * DM), *k16 = alloc16((size_t)TOK * DM);
  _Float16 *v16 = alloc16((size_t)TOK * DM), *ctx16 = alloc16((size_t)TOK * DM);
  _Float16 *o1h = alloc16((size_t)TOK * DM), *o2h = alloc16((size_t)TOK * DM);
  _Float16 *h1h = alloc16((size_t)TOK * DF);
  float *t32 = alloc32((size_t)TOK * DM);
  float *o1f = alloc32((size_t)TOK * DM);
  float *o2f = alloc32((size_t)TOK * DM);

  auto cvt = [&](const float* s, _Float16* d, size_t n) {
    f32_to_f16_kernel<<<dim3((unsigned)((n + 255) / 256)), dim3(256), 0, stream>>>(s, d, (int)n);
  };
  // C16-only GEMM (projections / FFN1)
  auto gemm16 = [&](const _Float16* A, const _Float16* Bm, const float* bias,
                    _Float16* C16, int M, int N, int K) {
    gemm_wmma_kernel<false, false, true><<<dim3(N / 128, M / 128), dim3(256), 0, stream>>>(
        A, Bm, bias, nullptr, C16, M, N, K);
  };
  auto gemm16_relu = [&](const _Float16* A, const _Float16* Bm, const float* bias,
                         _Float16* C16, int M, int N, int K) {
    gemm_wmma_kernel<true, false, true><<<dim3(N / 128, M / 128), dim3(256), 0, stream>>>(
        A, Bm, bias, nullptr, C16, M, N, K);
  };
  // C32-only GEMM (output projections / FFN2)
  auto gemm32 = [&](const _Float16* A, const _Float16* Bm, const float* bias,
                    float* C32, int M, int N, int K) {
    gemm_wmma_kernel<false, true, false><<<dim3(N / 128, M / 128), dim3(256), 0, stream>>>(
        A, Bm, bias, C32, nullptr, M, N, K);
  };

  // ---- converts (weights + activations) ----
  cvt(x, x16, (size_t)TOK * DM);
  cvt(enc, e16, (size_t)TOK * DM);
  cvt(wq1, hq1, (size_t)DM * DM); cvt(wk1, hk1, (size_t)DM * DM);
  cvt(wv1, hv1, (size_t)DM * DM); cvt(wo1, ho1, (size_t)DM * DM);
  cvt(wq2, hq2, (size_t)DM * DM); cvt(wk2, hk2, (size_t)DM * DM);
  cvt(wv2, hv2, (size_t)DM * DM); cvt(wo2, ho2, (size_t)DM * DM);
  cvt(fw1, hw1, (size_t)DM * DF); cvt(fw2, hw2, (size_t)DF * DM);

  // ---- MHA1: causal self-attention ----
  gemm16(x16, hq1, bq1, q16, TOK, DM, DM);
  gemm16(x16, hk1, bk1, k16, TOK, DM, DM);
  gemm16(x16, hv1, bv1, v16, TOK, DM, DM);
  flash_attn_kernel<<<dim3(SEQ / 64, NH, NB), dim3(128), 0, stream>>>(
      q16, k16, v16, nullptr, ctx16, 1);
  gemm32(ctx16, ho1, bo1, t32, TOK, DM, DM);
  layernorm_kernel<true><<<dim3(TOK), dim3(256), 0, stream>>>(t32, x, g1, be1, o1f, o1h);

  // ---- MHA2: cross-attention (q from out1, k/v from encoder) ----
  gemm16(o1h, hq2, bq2, q16, TOK, DM, DM);
  gemm16(e16, hk2, bk2, k16, TOK, DM, DM);
  gemm16(e16, hv2, bv2, v16, TOK, DM, DM);
  flash_attn_kernel<<<dim3(SEQ / 64, NH, NB), dim3(128), 0, stream>>>(
      q16, k16, v16, padm, ctx16, 0);
  gemm32(ctx16, ho2, bo2, t32, TOK, DM, DM);
  layernorm_kernel<true><<<dim3(TOK), dim3(256), 0, stream>>>(t32, o1f, g2, be2, o2f, o2h);

  // ---- FFN ----
  gemm16_relu(o2h, hw1, fb1, h1h, TOK, DF, DM);
  gemm32(h1h, hw2, fb2, t32, TOK, DM, DF);
  layernorm_kernel<false><<<dim3(TOK), dim3(256), 0, stream>>>(
      t32, o2f, g3, be3, (float*)d_out, nullptr);
}
